// THEGCNModel_24816321036923
// MI455X (gfx1250) — matrix operand order repaired and checked by reading, not compiled, not run on hardware
//
#include <hip/hip_runtime.h>
#include <hip/hip_bf16.h>
#include <math.h>

typedef _Float16 v16h __attribute__((ext_vector_type(16)));
typedef _Float16 v8h  __attribute__((ext_vector_type(8)));
typedef _Float16 v4h  __attribute__((ext_vector_type(4)));
typedef float    v8f  __attribute__((ext_vector_type(8)));

#define BN_EPS 1e-5f

// ---------------------------------------------------------------- zero
__global__ void k_zero(float* __restrict__ p, long n) {
    long i = (long)blockIdx.x * blockDim.x + threadIdx.x;
    long stride = (long)gridDim.x * blockDim.x;
    for (; i < n; i += stride) p[i] = 0.f;
}

// ---------------------------------------------------------------- pack smp weights into WMMA B-fragment order (f16)
// P1 layout: [l][nt(4)][ks(4)][lane(32)][j(16)]  (8192 halves / layer)
// P2 layout: [l][nt(4)][ks(2)][lane(32)][j(16)]  (4096 halves / layer)
__global__ void k_pack(const float* __restrict__ W1, const float* __restrict__ W2,
                       _Float16* __restrict__ P1, _Float16* __restrict__ P2, int L) {
    int idx = blockIdx.x * blockDim.x + threadIdx.x;
    int n1 = L * 8192;
    int n2 = L * 4096;
    if (idx < n1) {
        int j = idx & 15, lane = (idx >> 4) & 31, ks = (idx >> 9) & 3, nt = (idx >> 11) & 3, l = idx >> 13;
        int col = 16 * nt + (lane & 15);
        int k   = 32 * ks + 8 * (lane >> 4) + (j & 7) + ((j >> 3) << 4);
        P1[idx] = (_Float16)W1[((long)l * 128 + k) * 64 + col];
    }
    if (idx < n2) {
        int j = idx & 15, lane = (idx >> 4) & 31, ks = (idx >> 9) & 1, nt = (idx >> 10) & 3, l = idx >> 12;
        int col = 16 * nt + (lane & 15);
        int k   = 32 * ks + 8 * (lane >> 4) + (j & 7) + ((j >> 3) << 4);
        P2[idx] = (_Float16)W2[((long)l * 64 + k) * 64 + col];
    }
}

// ---------------------------------------------------------------- stage-1 temporal edge MLP (scalar, 66->33->17)
__global__ __launch_bounds__(256) void k_edge_stage1(
    const float* __restrict__ x, const int* __restrict__ ei, const float* __restrict__ dts,
    const float* __restrict__ freq, const float* __restrict__ phase,
    const float* __restrict__ W1, const float* __restrict__ b1,
    const float* __restrict__ W2, const float* __restrict__ b2,
    float* __restrict__ sum17, float* __restrict__ cnt, int E)
{
    __shared__ float sW1[66 * 33];
    __shared__ float sb1[33];
    __shared__ float sW2[33 * 17];
    __shared__ float sb2[17];
    __shared__ float sf[32], sp[32];
    __shared__ float h1L[33 * 256];
    for (int i = threadIdx.x; i < 66 * 33; i += 256) sW1[i] = W1[i];
    for (int i = threadIdx.x; i < 33 * 17; i += 256) sW2[i] = W2[i];
    if (threadIdx.x < 33) sb1[threadIdx.x] = b1[threadIdx.x];
    if (threadIdx.x < 17) sb2[threadIdx.x] = b2[threadIdx.x];
    if (threadIdx.x < 32) { sf[threadIdx.x] = freq[threadIdx.x]; sp[threadIdx.x] = phase[threadIdx.x]; }
    __syncthreads();

    int e = blockIdx.x * 256 + threadIdx.x;
    bool valid = (e < E);
    int ec = valid ? e : 0;
    int s = ei[ec];
    int d = ei[(long)E + ec];
    float dt = dts[ec];

    float xd[17], xs[17], rl[32];
#pragma unroll
    for (int i = 0; i < 17; i++) { xd[i] = x[(long)d * 17 + i]; xs[i] = x[(long)s * 17 + i]; }
#pragma unroll
    for (int i = 0; i < 32; i++) rl[i] = __cosf(dt * sf[i] + sp[i]);

    for (int j = 0; j < 33; j++) {
        float acc = sb1[j];
#pragma unroll
        for (int i = 0; i < 17; i++) acc += xd[i] * sW1[i * 33 + j];
#pragma unroll
        for (int i = 0; i < 17; i++) acc += xs[i] * sW1[(17 + i) * 33 + j];
#pragma unroll
        for (int i = 0; i < 32; i++) acc += rl[i] * sW1[(34 + i) * 33 + j];
        h1L[threadIdx.x + 256 * j] = fmaxf(acc, 0.f);
    }
    if (valid) {
        for (int o = 0; o < 17; o++) {
            float acc = sb2[o];
#pragma unroll
            for (int j = 0; j < 33; j++) acc += h1L[threadIdx.x + 256 * j] * sW2[j * 17 + o];
            float p = tanhf(acc);
            atomicAdd(&sum17[(long)d * 17 + o], (2.f * p - 1.f) * xd[o]);
        }
        atomicAdd(&cnt[d], 1.f);
    }
}

// ---------------------------------------------------------------- h = (x + mean) @ proj_W + proj_b   (17 -> 64)
__global__ __launch_bounds__(256) void k_proj(
    const float* __restrict__ x, const float* __restrict__ sum17, const float* __restrict__ cnt,
    const float* __restrict__ pW, const float* __restrict__ pb,
    float* __restrict__ h, int N)
{
    __shared__ float sW[17 * 64];
    __shared__ float sb[64];
    __shared__ float sh[4][17];
    for (int i = threadIdx.x; i < 17 * 64; i += 256) sW[i] = pW[i];
    if (threadIdx.x < 64) sb[threadIdx.x] = pb[threadIdx.x];
    int n0 = blockIdx.x * 4;
    if (threadIdx.x < 68) {
        int ni = threadIdx.x / 17, i = threadIdx.x % 17;
        int n = n0 + ni;
        if (n < N) {
            float c = fmaxf(cnt[n], 1.f);
            sh[ni][i] = x[(long)n * 17 + i] + sum17[(long)n * 17 + i] / c;
        }
    }
    __syncthreads();
    int ni = threadIdx.x >> 6, c = threadIdx.x & 63;
    int n = n0 + ni;
    if (n < N) {
        float acc = sb[c];
#pragma unroll
        for (int i = 0; i < 17; i++) acc += sh[ni][i] * sW[i * 64 + c];
        h[(long)n * 64 + c] = acc;
    }
}

// ---------------------------------------------------------------- stage-2 edge MLP, WMMA f16 (128->64->64), atomic scatter
// 8 waves / block, 16 edges / wave.
__global__ __launch_bounds__(256) void k_edge_wmma(
    const float* __restrict__ h, const int* __restrict__ ei, int E,
    const _Float16* __restrict__ P1, const _Float16* __restrict__ P2,
    const float* __restrict__ b1, const float* __restrict__ b2,
    float* __restrict__ sum64)
{
    __shared__ _Float16 feat[8][16][128];   // concat(h[dst], h[src]) per edge, f16
    __shared__ _Float16 hmid[8][16][64];    // relu(layer1) activations
    __shared__ int      dnode[8][16];

    int tid  = threadIdx.x;
    int w    = tid >> 5;
    int lane = tid & 31;
    int m    = lane & 15;   // edge slot within wave tile / matrix row group
    int hi   = lane >> 4;   // 0: dst-half, 1: src-half

    int base = blockIdx.x * 128 + w * 16;
    int e  = base + m;
    int ec = min(e, E - 1);
    int node = hi ? ei[ec] : ei[(long)E + ec];   // hi==0 -> dst (x_i), hi==1 -> src
    if (!hi) dnode[w][m] = node;

    // stage 64 features of this node as f16 into LDS (float4 global loads)
    const float4* hv = (const float4*)(h + (long)node * 64);
#pragma unroll
    for (int q = 0; q < 16; q++) {
        float4 v = hv[q];
        v4h p;
        p[0] = (_Float16)v.x; p[1] = (_Float16)v.y; p[2] = (_Float16)v.z; p[3] = (_Float16)v.w;
        *(v4h*)&feat[w][m][hi * 64 + q * 4] = p;
    }
    __syncthreads();

    // ---- GEMM1: [16 x 128] @ [128 x 64]
    v8f acc[4];
#pragma unroll
    for (int nt = 0; nt < 4; nt++) acc[nt] = (v8f){0.f,0.f,0.f,0.f,0.f,0.f,0.f,0.f};
#pragma unroll
    for (int ks = 0; ks < 4; ks++) {
        union { v16h v; v8h h8[2]; } a;
        // A-fragment (16-bit 16x32 layout): j0..7 -> K0 + 8*hi + j ; j8..15 -> K0 + 16 + 8*hi + (j-8)
        a.h8[0] = *(const v8h*)&feat[w][m][32 * ks + 8 * hi];
        a.h8[1] = *(const v8h*)&feat[w][m][32 * ks + 16 + 8 * hi];
#pragma unroll
        for (int nt = 0; nt < 4; nt++) {
            v16h b = *(const v16h*)&P1[(size_t)((nt * 4 + ks) * 32 + lane) * 16];
            acc[nt] = __builtin_amdgcn_wmma_f32_16x16x32_f16(false, a.v, false, b, (short)0, acc[nt], false, false);
        }
    }
    // bias + relu -> LDS as f16 for second GEMM
#pragma unroll
    for (int nt = 0; nt < 4; nt++) {
        int col = 16 * nt + m;
#pragma unroll
        for (int r = 0; r < 8; r++) {
            int row = hi ? (r + 8) : r;   // D layout: VGPR r -> M=r (lanes 0-15) / M=r+8 (lanes 16-31)
            float v = acc[nt][r] + b1[col];
            hmid[w][row][col] = (_Float16)fmaxf(v, 0.f);
        }
    }
    __syncthreads();

    // ---- GEMM2: [16 x 64] @ [64 x 64]
    v8f acc2[4];
#pragma unroll
    for (int nt = 0; nt < 4; nt++) acc2[nt] = (v8f){0.f,0.f,0.f,0.f,0.f,0.f,0.f,0.f};
#pragma unroll
    for (int ks = 0; ks < 2; ks++) {
        union { v16h v; v8h h8[2]; } a;
        a.h8[0] = *(const v8h*)&hmid[w][m][32 * ks + 8 * hi];
        a.h8[1] = *(const v8h*)&hmid[w][m][32 * ks + 16 + 8 * hi];
#pragma unroll
        for (int nt = 0; nt < 4; nt++) {
            v16h b = *(const v16h*)&P2[(size_t)((nt * 2 + ks) * 32 + lane) * 16];
            acc2[nt] = __builtin_amdgcn_wmma_f32_16x16x32_f16(false, a.v, false, b, (short)0, acc2[nt], false, false);
        }
    }

    // tanh gate, msg = (2p-1) * h_dst, coalesced atomic scatter into per-node sums
#pragma unroll
    for (int nt = 0; nt < 4; nt++) {
        int col = 16 * nt + m;
#pragma unroll
        for (int r = 0; r < 8; r++) {
            int row = hi ? (r + 8) : r;
            float p   = tanhf(acc2[nt][r] + b2[col]);
            float msg = (2.f * p - 1.f) * (float)feat[w][row][col];  // cols 0..63 = dst half
            int ee = base + row;
            if (ee < E) atomicAdd(&sum64[(long)dnode[w][row] * 64 + col], msg);
        }
    }
}

// ---------------------------------------------------------------- node update + BN stats accumulation
__global__ __launch_bounds__(256) void k_node_update(
    const float* __restrict__ hin, const float* __restrict__ sum64, const float* __restrict__ cnt,
    float* __restrict__ hout, float* __restrict__ stat, int N)
{
    __shared__ float bs[64], bq[64];
    if (threadIdx.x < 64) { bs[threadIdx.x] = 0.f; bq[threadIdx.x] = 0.f; }
    __syncthreads();
    long idx = (long)blockIdx.x * 256 + threadIdx.x;
    int n = (int)(idx >> 6), c = (int)(idx & 63);
    if (n < N) {
        float v = hin[idx] + sum64[idx] / fmaxf(cnt[n], 1.f);
        hout[idx] = v;
        atomicAdd(&bs[c], v);
        atomicAdd(&bq[c], v * v);
    }
    __syncthreads();
    if (threadIdx.x < 64) {
        atomicAdd(&stat[threadIdx.x], bs[threadIdx.x]);
        atomicAdd(&stat[64 + threadIdx.x], bq[threadIdx.x]);
    }
}

// ---------------------------------------------------------------- apply BN + relu in place
__global__ void k_bn_relu(float* __restrict__ hv, const float* __restrict__ stat,
                          const float* __restrict__ g, const float* __restrict__ b, int N)
{
    long idx = (long)blockIdx.x * 256 + threadIdx.x;
    if (idx < (long)N * 64) {
        int c = (int)(idx & 63);
        float inv  = 1.f / (float)N;
        float mean = stat[c] * inv;
        float var  = stat[64 + c] * inv - mean * mean;
        float v = (hv[idx] - mean) * rsqrtf(var + BN_EPS) * g[c] + b[c];
        hv[idx] = fmaxf(v, 0.f);
    }
}

// ---------------------------------------------------------------- classifier stage 1: z1 = h @ W1 + b1 (+stats)
__global__ __launch_bounds__(256) void k_clf1(
    const float* __restrict__ hfin, const float* __restrict__ W, const float* __restrict__ b,
    float* __restrict__ z1, float* __restrict__ stat, int rows)
{
    __shared__ float sW[64 * 64];
    __shared__ float sr[4][64];
    __shared__ float bs[64], bq[64];
    for (int i = threadIdx.x; i < 64 * 64; i += 256) sW[i] = W[i];
    if (threadIdx.x < 64) { bs[threadIdx.x] = 0.f; bq[threadIdx.x] = 0.f; }
    int r0 = blockIdx.x * 4;
    {
        int ri = threadIdx.x >> 6, c = threadIdx.x & 63;
        int r = r0 + ri;
        sr[ri][c] = (r < rows) ? hfin[(long)r * 64 + c] : 0.f;
    }
    __syncthreads();
    int ri = threadIdx.x >> 6, c = threadIdx.x & 63;
    int r = r0 + ri;
    float acc = b[c];
#pragma unroll
    for (int j = 0; j < 64; j++) acc += sr[ri][j] * sW[j * 64 + c];
    if (r < rows) {
        z1[(long)r * 64 + c] = acc;
        atomicAdd(&bs[c], acc); atomicAdd(&bq[c], acc * acc);
    }
    __syncthreads();
    if (threadIdx.x < 64) {
        atomicAdd(&stat[threadIdx.x], bs[threadIdx.x]);
        atomicAdd(&stat[64 + threadIdx.x], bq[threadIdx.x]);
    }
}

// ---------------------------------------------------------------- classifier stage 2: z2 = relu(bn(z1)) @ W2 + b2 (+stats)
__global__ __launch_bounds__(256) void k_clf2(
    const float* __restrict__ z1, const float* __restrict__ stat1,
    const float* __restrict__ g1, const float* __restrict__ bb1,
    const float* __restrict__ W, const float* __restrict__ b,
    float* __restrict__ z2, float* __restrict__ stat2, int rows)
{
    __shared__ float sW[64 * 32];
    __shared__ float sr[8][64];
    __shared__ float bs[32], bq[32];
    for (int i = threadIdx.x; i < 64 * 32; i += 256) sW[i] = W[i];
    if (threadIdx.x < 32) { bs[threadIdx.x] = 0.f; bq[threadIdx.x] = 0.f; }
    int r0 = blockIdx.x * 8;
    float inv = 1.f / (float)rows;
    for (int i = threadIdx.x; i < 8 * 64; i += 256) {
        int ri = i >> 6, c = i & 63;
        int r = r0 + ri;
        float v = 0.f;
        if (r < rows) {
            float mean = stat1[c] * inv;
            float var  = stat1[64 + c] * inv - mean * mean;
            v = fmaxf((z1[(long)r * 64 + c] - mean) * rsqrtf(var + BN_EPS) * g1[c] + bb1[c], 0.f);
        }
        sr[ri][c] = v;
    }
    __syncthreads();
    int ri = threadIdx.x >> 5, c = threadIdx.x & 31;
    int r = r0 + ri;
    float acc = b[c];
#pragma unroll
    for (int j = 0; j < 64; j++) acc += sr[ri][j] * sW[j * 32 + c];
    if (r < rows) {
        z2[(long)r * 32 + c] = acc;
        atomicAdd(&bs[c], acc); atomicAdd(&bq[c], acc * acc);
    }
    __syncthreads();
    if (threadIdx.x < 32) {
        atomicAdd(&stat2[threadIdx.x], bs[threadIdx.x]);
        atomicAdd(&stat2[32 + threadIdx.x], bq[threadIdx.x]);
    }
}

// ---------------------------------------------------------------- classifier stage 3: out = relu(bn(z2)) @ W3 + b3
__global__ void k_clf3(const float* __restrict__ z2, const float* __restrict__ stat2,
                       const float* __restrict__ g2, const float* __restrict__ bb2,
                       const float* __restrict__ W3, const float* __restrict__ b3,
                       float* __restrict__ out, int rows)
{
    int r = blockIdx.x * 256 + threadIdx.x;
    if (r >= rows) return;
    float inv = 1.f / (float)rows;
    float acc = b3[0];
#pragma unroll
    for (int c = 0; c < 32; c++) {
        float mean = stat2[c] * inv;
        float var  = stat2[32 + c] * inv - mean * mean;
        float v = fmaxf((z2[(long)r * 32 + c] - mean) * rsqrtf(var + BN_EPS) * g2[c] + bb2[c], 0.f);
        acc += v * W3[c];
    }
    out[r] = acc;
}

// ================================================================ launch
extern "C" void kernel_launch(void* const* d_in, const int* in_sizes, int n_in,
                              void* d_out, int out_size, void* d_ws, size_t ws_size,
                              hipStream_t stream) {
    const float* x       = (const float*)d_in[0];
    const int*   ei      = (const int*)d_in[1];
    const float* dts     = (const float*)d_in[2];
    const float* freq    = (const float*)d_in[4];
    const float* phase   = (const float*)d_in[5];
    const float* tmp_W1  = (const float*)d_in[6];
    const float* tmp_b1  = (const float*)d_in[7];
    const float* tmp_W2  = (const float*)d_in[8];
    const float* tmp_b2  = (const float*)d_in[9];
    const float* proj_W  = (const float*)d_in[10];
    const float* proj_b  = (const float*)d_in[11];
    const float* smp_W1  = (const float*)d_in[12];
    const float* smp_b1  = (const float*)d_in[13];
    const float* smp_W2  = (const float*)d_in[14];
    const float* smp_b2  = (const float*)d_in[15];
    const float* bn_g    = (const float*)d_in[16];
    const float* bn_b    = (const float*)d_in[17];
    const float* clf_W1  = (const float*)d_in[18];
    const float* clf_b1  = (const float*)d_in[19];
    const float* clf_g1  = (const float*)d_in[20];
    const float* clf_bb1 = (const float*)d_in[21];
    const float* clf_W2  = (const float*)d_in[22];
    const float* clf_b2  = (const float*)d_in[23];
    const float* clf_g2  = (const float*)d_in[24];
    const float* clf_bb2 = (const float*)d_in[25];
    const float* clf_W3  = (const float*)d_in[26];
    const float* clf_b3  = (const float*)d_in[27];
    float* out = (float*)d_out;

    int N    = in_sizes[0] / 17;
    int E    = in_sizes[2];
    int L    = in_sizes[13] / 64;      // smp_b1 is (L, 64)
    int rows = out_size;               // output is (batch_size, 1)

    // ---- carve workspace (bump allocator, 32B aligned slices)
    float* ws = (float*)d_ws;
    size_t off = 0;
    auto alloc = [&](size_t n) { float* p = ws + off; off += (n + 7) & ~(size_t)7; return p; };
    float* cnt    = alloc((size_t)N);
    float* sum17  = alloc((size_t)N * 17);
    float* sum64  = alloc((size_t)N * 64);
    float* stat   = alloc(128);
    float* hA     = alloc((size_t)N * 64);
    float* hB     = alloc((size_t)N * 64);
    float* z1pre  = alloc((size_t)rows * 64);
    float* z2pre  = alloc((size_t)rows * 32);
    float* stat1  = alloc(128);
    float* stat2  = alloc(64);
    _Float16* P1  = (_Float16*)alloc((size_t)L * 4096);  // L*8192 halves
    _Float16* P2  = (_Float16*)alloc((size_t)L * 2048);  // L*4096 halves

    const int ZB = 2048;  // grid-stride zero blocks

    // pack WMMA B fragments for both GNN layers
    {
        int npack = L * 8192;
        k_pack<<<(npack + 255) / 256, 256, 0, stream>>>(smp_W1, smp_W2, P1, P2, L);
    }

    // stage 1: temporal message passing (also produces in-degree counts, reused everywhere)
    k_zero<<<ZB, 256, 0, stream>>>(cnt, (long)N);
    k_zero<<<ZB, 256, 0, stream>>>(sum17, (long)N * 17);
    k_edge_stage1<<<(E + 255) / 256, 256, 0, stream>>>(x, ei, dts, freq, phase,
                                                       tmp_W1, tmp_b1, tmp_W2, tmp_b2,
                                                       sum17, cnt, E);
    k_proj<<<(N + 3) / 4, 256, 0, stream>>>(x, sum17, cnt, proj_W, proj_b, hA, N);

    // stage 2: WMMA edge MLP layers
    float* hcur = hA;
    float* hnxt = hB;
    for (int l = 0; l < L; l++) {
        k_zero<<<ZB, 256, 0, stream>>>(sum64, (long)N * 64);
        k_zero<<<1, 256, 0, stream>>>(stat, 128);
        k_edge_wmma<<<(E + 127) / 128, 256, 0, stream>>>(hcur, ei, E,
                                                         P1 + (size_t)l * 8192, P2 + (size_t)l * 4096,
                                                         smp_b1 + l * 64, smp_b2 + l * 64, sum64);
        k_node_update<<<(int)(((long)N * 64 + 255) / 256), 256, 0, stream>>>(hcur, sum64, cnt, hnxt, stat, N);
        k_bn_relu<<<(int)(((long)N * 64 + 255) / 256), 256, 0, stream>>>(hnxt, stat, bn_g + l * 64, bn_b + l * 64, N);
        float* t = hcur; hcur = hnxt; hnxt = t;
    }

    // classifier head on first `rows` nodes
    k_zero<<<1, 256, 0, stream>>>(stat1, 192);   // stat1(128) + stat2(64) are adjacent
    k_clf1<<<(rows + 3) / 4, 256, 0, stream>>>(hcur, clf_W1, clf_b1, z1pre, stat1, rows);
    k_clf2<<<(rows + 7) / 8, 256, 0, stream>>>(z1pre, stat1, clf_g1, clf_bb1, clf_W2, clf_b2,
                                               z2pre, stat2, rows);
    k_clf3<<<(rows + 255) / 256, 256, 0, stream>>>(z2pre, stat2, clf_g2, clf_bb2, clf_W3, clf_b3,
                                                   out, rows);
    (void)n_in; (void)ws_size; (void)d_in;
}
